// Encoder_45535243272203
// MI455X (gfx1250) — matrix operand order repaired and checked by reading, not compiled, not run on hardware
//
#include <hip/hip_runtime.h>
#include <stdint.h>

// ---------------------------------------------------------------------------
// Transformer encoder (B=16, T=1024, D=512, H=8, FF=2048, L=6) for gfx1250.
// All GEMMs + attention use v_wmma_f32_16x16x32_f16 (wave32 WMMA, f32 accum).
// ---------------------------------------------------------------------------

namespace {
constexpr int kB  = 16;
constexpr int kT  = 1024;
constexpr int kD  = 512;
constexpr int kH  = 8;
constexpr int kFF = 2048;
constexpr int kL  = 6;
constexpr int kM  = kB * kT;          // 16384 token rows
constexpr float kNeg   = -1e9f;
constexpr float kEps   = 1e-5f;
constexpr float kScale = 0.125f;      // 1/sqrt(DH=64)
}

typedef _Float16 f16;
typedef __attribute__((ext_vector_type(16))) _Float16 v16h;
typedef __attribute__((ext_vector_type(8)))  float    v8f;

union FragU {
  v16h  v;
  uint4 q[2];
};

__device__ __forceinline__ v8f wmma_f16(v16h a, v16h b, v8f c) {
  // D = A(16x32 f16) * B(32x16 f16) + C(16x16 f32)
  return __builtin_amdgcn_wmma_f32_16x16x32_f16(false, a, false, b, (short)0, c,
                                                false, false);
}

// A-operand fragment (16x32, row = lane&15). p -> element (row, k0).
// ISA layout: VGPR0-3 hold K = hi*8..hi*8+7, VGPR4-7 hold K = 16+hi*8..+7.
__device__ __forceinline__ v16h load_frag_a(const f16* p, int hi) {
  FragU f;
  f.q[0] = *(const uint4*)(p + hi * 8);
  f.q[1] = *(const uint4*)(p + 16 + hi * 8);
  return f.v;
}

// B-operand fragment (32x16, col = lane&15). p -> element (col, k0) of B^T
// stored row-major.  ISA layout: VGPR0-3 hold K = hi*8..hi*8+7, VGPR4-7 hold K = 16+hi*8..+7.
__device__ __forceinline__ v16h load_frag_b(const f16* p, int hi) {
  FragU f;
  f.q[0] = *(const uint4*)(p + hi * 8);
  f.q[1] = *(const uint4*)(p + 16 + hi * 8);
  return f.v;
}

// ---------------------------------------------------------------------------
// x = inputs + positional encoding (pe broadcast over batch)
// ---------------------------------------------------------------------------
__global__ __launch_bounds__(256) void add_pe_kernel(
    const float* __restrict__ in, const float* __restrict__ pe,
    float* __restrict__ x) {
  const int i   = blockIdx.x * 256 + threadIdx.x;  // over kM*kD
  const int row = i >> 9;                          // /512
  const int d   = i & 511;
  const int t   = row & (kT - 1);
  x[i] = in[i] + pe[t * kD + d];
}

// ---------------------------------------------------------------------------
// fp32 W[K,N] -> f16 W^T[N,K]
// ---------------------------------------------------------------------------
__global__ __launch_bounds__(256) void convert_transpose_kernel(
    const float* __restrict__ w, f16* __restrict__ wt, int K, int N) {
  const int i = blockIdx.x * 256 + threadIdx.x;
  if (i >= K * N) return;
  const int k = i / N, n = i % N;
  wt[(size_t)n * K + k] = (f16)w[i];
}

// ---------------------------------------------------------------------------
// LayerNorm: one wave32 per 512-element row, f16 output.
// ---------------------------------------------------------------------------
__global__ __launch_bounds__(256) void layernorm_kernel(
    const float* __restrict__ x, const float* __restrict__ g,
    const float* __restrict__ bta, f16* __restrict__ h) {
  const int w = threadIdx.x >> 5, lane = threadIdx.x & 31;
  const int row  = blockIdx.x * 8 + w;
  const int base = lane * 16;
  const float* xr = x + (size_t)row * kD;
  float v[16];
  float s = 0.f;
#pragma unroll
  for (int i = 0; i < 16; ++i) { v[i] = xr[base + i]; s += v[i]; }
#pragma unroll
  for (int m = 16; m >= 1; m >>= 1) s += __shfl_xor(s, m, 32);
  const float mu = s * (1.f / kD);
  float q = 0.f;
#pragma unroll
  for (int i = 0; i < 16; ++i) { const float d = v[i] - mu; q += d * d; }
#pragma unroll
  for (int m = 16; m >= 1; m >>= 1) q += __shfl_xor(q, m, 32);
  const float rs = rsqrtf(q * (1.f / kD) + kEps);
  f16* hr = h + (size_t)row * kD;
#pragma unroll
  for (int i = 0; i < 16; ++i) {
    const int c = base + i;
    hr[c] = (f16)((v[i] - mu) * rs * g[c] + bta[c]);
  }
}

// ---------------------------------------------------------------------------
// WMMA GEMM: C[M,N] = A[M,K] (f16, row-major) * Bt[N,K]^T + bias.
// 8 waves/block; block tile 128x128; wave tile 32x64 (8 accumulators).
// mode 0: store f16                       (Q, K projections)
// mode 1: store f16, relu                 (FFN1)
// mode 2: fp32 residual accumulate into x (FFN2)
// mode 3: store f16 transposed per head -> V^T [B,H,DH,T] (V projection)
// ---------------------------------------------------------------------------
__global__ __launch_bounds__(256) void gemm_wmma_kernel(
    const f16* __restrict__ A, const f16* __restrict__ Bt,
    const float* __restrict__ bias, void* __restrict__ outp,
    int K, int N, int mode) {
  const int tid  = threadIdx.x;
  const int w    = tid >> 5;
  const int lane = tid & 31;
  const int r    = lane & 15;
  const int hi   = lane >> 4;
  const int wm   = w & 3;
  const int wn   = w >> 2;
  const int rowBase = blockIdx.y * 128 + wm * 32;
  const int colBase = blockIdx.x * 128 + wn * 64;

  const v8f z = {0.f, 0.f, 0.f, 0.f, 0.f, 0.f, 0.f, 0.f};
  v8f acc[2][4];
#pragma unroll
  for (int i = 0; i < 2; ++i)
#pragma unroll
    for (int j = 0; j < 4; ++j) acc[i][j] = z;

  const f16* a0p = A + (size_t)(rowBase + r) * K;
  const f16* a1p = A + (size_t)(rowBase + 16 + r) * K;
  const f16* bp[4];
#pragma unroll
  for (int n0 = 0; n0 < 4; ++n0)
    bp[n0] = Bt + (size_t)(colBase + n0 * 16 + r) * K;

  for (int kk = 0; kk < K; kk += 32) {
    const v16h af0 = load_frag_a(a0p + kk, hi);
    const v16h af1 = load_frag_a(a1p + kk, hi);
#pragma unroll
    for (int n0 = 0; n0 < 4; ++n0) {
      const v16h bf = load_frag_b(bp[n0] + kk, hi);
      acc[0][n0] = wmma_f16(af0, bf, acc[0][n0]);
      acc[1][n0] = wmma_f16(af1, bf, acc[1][n0]);
    }
  }

#pragma unroll
  for (int m0 = 0; m0 < 2; ++m0) {
#pragma unroll
    for (int n0 = 0; n0 < 4; ++n0) {
      const int   col = colBase + n0 * 16 + r;
      const float bia = bias[col];
#pragma unroll
      for (int j = 0; j < 8; ++j) {
        const int row = rowBase + m0 * 16 + hi * 8 + j;  // C layout: M=j+8*hi
        float v = acc[m0][n0][j] + bia;
        if (mode == 1) v = fmaxf(v, 0.f);
        if (mode == 2) {
          ((float*)outp)[(size_t)row * N + col] += v;
        } else if (mode == 3) {
          const int b = row >> 10, t = row & (kT - 1);
          const int h = col >> 6,  d = col & 63;
          ((f16*)outp)[(((size_t)(b * kH + h)) * 64 + d) * kT + t] = (f16)v;
        } else {
          ((f16*)outp)[(size_t)row * N + col] = (f16)v;
        }
      }
    }
  }
}

// ---------------------------------------------------------------------------
// Flash attention: one wave per 16-query tile of one (b,h); streams 32 keys
// per iteration, online softmax, LDS bounce to remap P (C-frag -> A-frag).
// Adds ctx into x (residual).
// ---------------------------------------------------------------------------
__global__ __launch_bounds__(256) void flash_attn_kernel(
    const f16* __restrict__ qb, const f16* __restrict__ kb,
    const f16* __restrict__ vt, const int* __restrict__ lens,
    float* __restrict__ x) {
  __shared__ __align__(16) f16 P[8][16 * 32];  // 1KB per wave
  const int w = threadIdx.x >> 5, lane = threadIdx.x & 31;
  const int r = lane & 15, hi = lane >> 4;
  const int bh = blockIdx.x;
  const int b = bh >> 3, h = bh & 7;
  const int t0  = (blockIdx.y * 8 + w) * 16;
  const int len = lens[b];

  const f16* qrow = qb + ((size_t)(b * kT + t0 + r)) * kD + h * 64;
  const v16h qf0 = load_frag_a(qrow, hi);
  const v16h qf1 = load_frag_a(qrow + 32, hi);

  float mst[8], lst[8];
#pragma unroll
  for (int j = 0; j < 8; ++j) { mst[j] = -3.0e38f; lst[j] = 0.f; }
  const v8f z = {0.f, 0.f, 0.f, 0.f, 0.f, 0.f, 0.f, 0.f};
  v8f o[4] = {z, z, z, z};

  f16* Pw = P[w];
  const f16* kbase = kb + ((size_t)b * kT) * kD + h * 64;
  const f16* vbase = vt + ((size_t)(b * kH + h)) * 64 * kT;  // [64][T]

  for (int kt = 0; kt < kT; kt += 32) {
    v8f s0 = z, s1 = z;
    {  // keys kt .. kt+15 (cols of S-tile 0)
      const f16* kr = kbase + (size_t)(kt + r) * kD;
      s0 = wmma_f16(qf0, load_frag_b(kr, hi), s0);
      s0 = wmma_f16(qf1, load_frag_b(kr + 32, hi), s0);
    }
    {  // keys kt+16 .. kt+31
      const f16* kr = kbase + (size_t)(kt + 16 + r) * kD;
      s1 = wmma_f16(qf0, load_frag_b(kr, hi), s1);
      s1 = wmma_f16(qf1, load_frag_b(kr + 32, hi), s1);
    }
    const bool msk0 = (kt + r) >= len;        // pad-mask per key column
    const bool msk1 = (kt + 16 + r) >= len;
#pragma unroll
    for (int j = 0; j < 8; ++j) {
      const float a = msk0 ? kNeg : s0[j] * kScale;
      const float c = msk1 ? kNeg : s1[j] * kScale;
      float mx = fmaxf(a, c);                 // row reduction across 16 lanes
      mx = fmaxf(mx, __shfl_xor(mx, 1, 32));
      mx = fmaxf(mx, __shfl_xor(mx, 2, 32));
      mx = fmaxf(mx, __shfl_xor(mx, 4, 32));
      mx = fmaxf(mx, __shfl_xor(mx, 8, 32));
      const float mnew = fmaxf(mst[j], mx);
      const float corr = __expf(mst[j] - mnew);
      const float e0 = __expf(a - mnew);
      const float e1 = __expf(c - mnew);
      float rsum = e0 + e1;
      rsum += __shfl_xor(rsum, 1, 32);
      rsum += __shfl_xor(rsum, 2, 32);
      rsum += __shfl_xor(rsum, 4, 32);
      rsum += __shfl_xor(rsum, 8, 32);
      lst[j] = lst[j] * corr + rsum;
      mst[j] = mnew;
      o[0][j] *= corr; o[1][j] *= corr; o[2][j] *= corr; o[3][j] *= corr;
      Pw[(hi * 8 + j) * 32 + r]      = (f16)e0;  // C-frag row = j+8*hi
      Pw[(hi * 8 + j) * 32 + 16 + r] = (f16)e1;
    }
    asm volatile("s_wait_dscnt 0x0" ::: "memory");  // per-wave LDS ordering
    v16h pf;  // re-read P as an A-operand fragment (row = r)
    {
      FragU f;
      f.q[0] = *(const uint4*)(&Pw[r * 32 + hi * 8]);
      f.q[1] = *(const uint4*)(&Pw[r * 32 + 16 + hi * 8]);
      pf = f.v;
    }
#pragma unroll
    for (int n0 = 0; n0 < 4; ++n0) {   // O += P (16x32) * V (32x64)
      const f16* vr = vbase + (size_t)(n0 * 16 + r) * kT + kt;
      o[n0] = wmma_f16(pf, load_frag_b(vr, hi), o[n0]);
    }
  }

#pragma unroll
  for (int n0 = 0; n0 < 4; ++n0) {
    const int col = h * 64 + n0 * 16 + r;
#pragma unroll
    for (int j = 0; j < 8; ++j) {
      const int row = b * kT + t0 + hi * 8 + j;
      x[(size_t)row * kD + col] += o[n0][j] / lst[j];
    }
  }
}

// ---------------------------------------------------------------------------
extern "C" void kernel_launch(void* const* d_in, const int* in_sizes, int n_in,
                              void* d_out, int out_size, void* d_ws,
                              size_t ws_size, hipStream_t stream) {
  (void)in_sizes; (void)n_in; (void)out_size; (void)ws_size;
  const float* inputs = (const float*)d_in[0];
  const int*   lens   = (const int*)d_in[1];
  const float* pe     = (const float*)d_in[2];
  const float* ln1g   = (const float*)d_in[3];
  const float* ln1b   = (const float*)d_in[4];
  const float* Wq     = (const float*)d_in[5];
  const float* bq     = (const float*)d_in[6];
  const float* Wk     = (const float*)d_in[7];
  const float* bk     = (const float*)d_in[8];
  const float* Wv     = (const float*)d_in[9];
  const float* bv     = (const float*)d_in[10];
  const float* ln2g   = (const float*)d_in[11];
  const float* ln2b   = (const float*)d_in[12];
  const float* W1     = (const float*)d_in[13];
  const float* b1     = (const float*)d_in[14];
  const float* W2     = (const float*)d_in[15];
  const float* b2     = (const float*)d_in[16];

  char* ws = (char*)d_ws;
  size_t off = 0;
  auto take = [&](size_t bytes) -> void* {
    void* p = ws + off;
    off += (bytes + 255) & ~(size_t)255;
    return p;
  };
  // ~174 MB total scratch
  float* x   = (float*)take((size_t)kM * kD * 4);    // activations fp32
  f16* hb    = (f16*)take((size_t)kM * kD * 2);      // LN output
  f16* qbuf  = (f16*)take((size_t)kM * kD * 2);
  f16* kbuf  = (f16*)take((size_t)kM * kD * 2);
  f16* vtb   = (f16*)take((size_t)kM * kD * 2);      // V^T [B,H,DH,T]
  f16* fbuf  = (f16*)take((size_t)kM * kFF * 2);     // FFN intermediate
  f16* wqT   = (f16*)take((size_t)kD * kD * 2);
  f16* wkT   = (f16*)take((size_t)kD * kD * 2);
  f16* wvT   = (f16*)take((size_t)kD * kD * 2);
  f16* w1T   = (f16*)take((size_t)kD * kFF * 2);
  f16* w2T   = (f16*)take((size_t)kFF * kD * 2);

  add_pe_kernel<<<(kM * kD) / 256, 256, 0, stream>>>(inputs, pe, x);

  const dim3 blk(256);
  const dim3 gD(kD / 128, kM / 128);     // N=512 gemms
  const dim3 gF(kFF / 128, kM / 128);    // N=2048 gemm
  const dim3 gA(kB * kH, kT / 128);      // attention: 128 (b,h) x 8 tiles

  for (int l = 0; l < kL; ++l) {
    convert_transpose_kernel<<<(kD * kD) / 256, blk, 0, stream>>>(
        Wq + (size_t)l * kD * kD, wqT, kD, kD);
    convert_transpose_kernel<<<(kD * kD) / 256, blk, 0, stream>>>(
        Wk + (size_t)l * kD * kD, wkT, kD, kD);
    convert_transpose_kernel<<<(kD * kD) / 256, blk, 0, stream>>>(
        Wv + (size_t)l * kD * kD, wvT, kD, kD);
    convert_transpose_kernel<<<(kD * kFF) / 256, blk, 0, stream>>>(
        W1 + (size_t)l * kD * kFF, w1T, kD, kFF);
    convert_transpose_kernel<<<(kFF * kD) / 256, blk, 0, stream>>>(
        W2 + (size_t)l * kFF * kD, w2T, kFF, kD);

    layernorm_kernel<<<kM / 8, blk, 0, stream>>>(x, ln1g + l * kD,
                                                 ln1b + l * kD, hb);
    gemm_wmma_kernel<<<gD, blk, 0, stream>>>(hb, wqT, bq + l * kD, qbuf,
                                             kD, kD, 0);
    gemm_wmma_kernel<<<gD, blk, 0, stream>>>(hb, wkT, bk + l * kD, kbuf,
                                             kD, kD, 0);
    gemm_wmma_kernel<<<gD, blk, 0, stream>>>(hb, wvT, bv + l * kD, vtb,
                                             kD, kD, 3);
    flash_attn_kernel<<<gA, blk, 0, stream>>>(qbuf, kbuf, vtb, lens, x);

    layernorm_kernel<<<kM / 8, blk, 0, stream>>>(x, ln2g + l * kD,
                                                 ln2b + l * kD, hb);
    gemm_wmma_kernel<<<gF, blk, 0, stream>>>(hb, w1T, b1 + l * kFF, fbuf,
                                             kD, kFF, 1);
    gemm_wmma_kernel<<<gD, blk, 0, stream>>>(fbuf, w2T, b2 + l * kD, x,
                                             kFF, kD, 2);
  }

  hipMemcpyAsync(d_out, x, (size_t)kM * kD * 4, hipMemcpyDeviceToDevice,
                 stream);
}
